// RiemannianAttention_31181462569466
// MI455X (gfx1250) — compile-verified, matrix operand order checked by reading, not run on hardware
//
#include <hip/hip_runtime.h>

typedef __attribute__((ext_vector_type(16))) _Float16 v16h;
typedef __attribute__((ext_vector_type(8)))  float    v8f;

#define S_LEN 2048
#define D_DIM 64
#define BH    32            // B*H
#define QTILE 128           // q rows per block (8 waves * 16)
#define KTILE 32            // keys per iteration
#define NKT   (S_LEN / KTILE)
#define KS_STRIDE 72        // padded halves per K row  (144B, conflict-free, 16B aligned)
#define VT_STRIDE 40        // padded halves per V^T row (80B, conflict-free, 16B aligned)
#define P_STRIDE  40        // padded halves per P row
#define LN2F 0.6931471805599453f

union V16 { v16h h; uint4 u[2]; };

__device__ __forceinline__ v8f wmma_f16(const V16& a, const V16& b, v8f c) {
  // D = A(16x32 f16) * B(32x16 f16) + C(16x16 f32)
  return __builtin_amdgcn_wmma_f32_16x16x32_f16(false, a.h, false, b.h,
                                                (short)0, c, false, false);
}

__global__ __launch_bounds__(256)
void poincare_flash_attn(const float* __restrict__ Q,
                         const float* __restrict__ K,
                         const float* __restrict__ V,
                         float* __restrict__ O) {
  __shared__ __align__(16) _Float16 k_s[2][KTILE * KS_STRIDE];   // K tiles, f16 row-major
  __shared__ __align__(16) _Float16 vt_s[2][D_DIM * VT_STRIDE];  // V tiles transposed [d][key]
  __shared__ __align__(16) _Float16 p_s[8 * 16 * P_STRIDE];      // per-wave P staging
  __shared__ float kn_s[2][KTILE];                               // exact f32 key norms

  const int t    = threadIdx.x;
  const int wave = t >> 5;
  const int lane = t & 31;
  const int col  = lane & 15;
  const int hi   = lane >> 4;
  const int hi8  = hi << 3;

  const int bh = blockIdx.x / (S_LEN / QTILE);
  const int q0 = (blockIdx.x % (S_LEN / QTILE)) * QTILE;
  const size_t base = (size_t)bh * S_LEN * D_DIM;
  const float* Qg = Q + base;
  const float* Kg = K + base;
  const float* Vg = V + base;
  float*       Og = O + base;

  // ---- load Q tile into A-layout f16 regs; exact f32 row norms ----
  // A 16x32 layout: lane row M = lane&15; element j <-> k = (j<8 ? j : j+8) + hi8
  V16 aq0, aq1;
  const int   qrow = q0 + wave * 16 + col;
  const float* qp  = Qg + (size_t)qrow * D_DIM;
  float qn_part = 0.f;
  #pragma unroll
  for (int j = 0; j < 16; ++j) {
    const int k  = ((j < 8) ? j : j + 8) + hi8;
    const float a = qp[k];
    const float b = qp[32 + k];
    aq0.h[j] = (_Float16)a;
    aq1.h[j] = (_Float16)b;
    qn_part += a * a + b * b;
  }
  const float qn = qn_part + __shfl_xor(qn_part, 16, 32);
  float qn_m[8], q1_m[8];                  // qn / (1-qn) for C-layout row m = i + hi8
  #pragma unroll
  for (int i = 0; i < 8; ++i) { qn_m[i] = __shfl(qn, i + hi8, 32); q1_m[i] = 1.f - qn_m[i]; }

  // ---- flash-attention running state, log2 domain (row m = i + hi8) ----
  float mrow[8], lrow[8];
  v8f o0 = {}, o1 = {}, o2 = {}, o3 = {};
  #pragma unroll
  for (int i = 0; i < 8; ++i) { mrow[i] = -3.0e38f; lrow[i] = 0.f; }

  const int srow = t >> 3;         // staging: key row 0..31
  const int sd0  = (t & 7) * 8;    // staging: 8 consecutive d

  // ---- preload tile 0 into registers (software pipeline) ----
  float4 kr0, kr1, vr0, vr1;
  {
    const float4* kp = reinterpret_cast<const float4*>(Kg + (size_t)srow * D_DIM + sd0);
    kr0 = kp[0]; kr1 = kp[1];
    const float4* vp = reinterpret_cast<const float4*>(Vg + (size_t)srow * D_DIM + sd0);
    vr0 = vp[0]; vr1 = vp[1];
  }

  for (int kt = 0; kt < NKT; ++kt) {
    const int cur = kt & 1;
    // ======= flush staged registers -> LDS buf[cur]; exact key norms ========
    {
      float part = kr0.x*kr0.x + kr0.y*kr0.y + kr0.z*kr0.z + kr0.w*kr0.w
                 + kr1.x*kr1.x + kr1.y*kr1.y + kr1.z*kr1.z + kr1.w*kr1.w;
      #pragma unroll
      for (int off = 1; off < 8; off <<= 1) part += __shfl_xor(part, off, 32);
      if ((t & 7) == 0) kn_s[cur][srow] = part;
      union { _Float16 h[8]; uint4 u; } pk;
      pk.h[0]=(_Float16)kr0.x; pk.h[1]=(_Float16)kr0.y; pk.h[2]=(_Float16)kr0.z; pk.h[3]=(_Float16)kr0.w;
      pk.h[4]=(_Float16)kr1.x; pk.h[5]=(_Float16)kr1.y; pk.h[6]=(_Float16)kr1.z; pk.h[7]=(_Float16)kr1.w;
      *reinterpret_cast<uint4*>(&k_s[cur][srow * KS_STRIDE + sd0]) = pk.u;
      _Float16* vt = vt_s[cur];
      vt[(sd0 + 0) * VT_STRIDE + srow] = (_Float16)vr0.x;
      vt[(sd0 + 1) * VT_STRIDE + srow] = (_Float16)vr0.y;
      vt[(sd0 + 2) * VT_STRIDE + srow] = (_Float16)vr0.z;
      vt[(sd0 + 3) * VT_STRIDE + srow] = (_Float16)vr0.w;
      vt[(sd0 + 4) * VT_STRIDE + srow] = (_Float16)vr1.x;
      vt[(sd0 + 5) * VT_STRIDE + srow] = (_Float16)vr1.y;
      vt[(sd0 + 6) * VT_STRIDE + srow] = (_Float16)vr1.z;
      vt[(sd0 + 7) * VT_STRIDE + srow] = (_Float16)vr1.w;
    }
    __syncthreads();   // single barrier per tile: buf[cur] visible, buf[cur^1] free

    // ======= issue next tile's global loads; latency hidden by compute ======
    if (kt + 1 < NKT) {
      const float4* kp = reinterpret_cast<const float4*>(
          Kg + ((size_t)((kt + 1) * KTILE + srow)) * D_DIM + sd0);
      kr0 = kp[0]; kr1 = kp[1];
      const float4* vp = reinterpret_cast<const float4*>(
          Vg + ((size_t)((kt + 1) * KTILE + srow)) * D_DIM + sd0);
      vr0 = vp[0]; vr1 = vp[1];
      if (kt + 2 < NKT) {   // global_prefetch_b8 at distance 2
        __builtin_prefetch(Kg + ((size_t)((kt + 2) * KTILE + srow)) * D_DIM + sd0, 0, 1);
        __builtin_prefetch(Vg + ((size_t)((kt + 2) * KTILE + srow)) * D_DIM + sd0, 0, 1);
      }
    }

    // ================= S = Q K^T : two 16x16 tiles, K-dim 64 ================
    v8f c0 = {}, c1 = {};
    {
      // B 32x16 layout: lane col N = lane&15; element j <-> d-row k = (j<8?j:j+8)+hi8
      V16 b00, b01, b10, b11;
      const _Float16* bp0 = &k_s[cur][(0  + col) * KS_STRIDE + hi8];
      b00.u[0] = *(const uint4*)(bp0);      b00.u[1] = *(const uint4*)(bp0 + 16);
      b01.u[0] = *(const uint4*)(bp0 + 32); b01.u[1] = *(const uint4*)(bp0 + 48);
      const _Float16* bp1 = &k_s[cur][(16 + col) * KS_STRIDE + hi8];
      b10.u[0] = *(const uint4*)(bp1);      b10.u[1] = *(const uint4*)(bp1 + 16);
      b11.u[0] = *(const uint4*)(bp1 + 32); b11.u[1] = *(const uint4*)(bp1 + 48);
      c0 = wmma_f16(aq0, b00, c0);  c0 = wmma_f16(aq1, b01, c0);
      c1 = wmma_f16(aq0, b10, c1);  c1 = wmma_f16(aq1, b11, c1);
    }

    // ==== Poincare distance -> scores in log2 units (rcp + log2, no div) ====
    const float kn0 = kn_s[cur][col];
    const float kn1 = kn_s[cur][16 + col];
    const float kc0 = 1.f - kn0;
    const float kc1 = 1.f - kn1;
    float s0[8], s1[8];
    #pragma unroll
    for (int i = 0; i < 8; ++i) {
      {
        const float diff2 = fmaxf(qn_m[i] + kn0 - 2.f * c0[i], 0.f);
        const float rden  = __builtin_amdgcn_rcpf(fmaxf(q1_m[i] * kc0, 1e-7f));
        float arg = fmaf(diff2 + diff2, rden, 1.f);
        arg = fmaxf(arg, 1.f + 1e-7f);
        const float L = __log2f(arg + __fsqrt_rn(arg * arg - 1.f));  // arccosh/ln2
        s0[i] = -(L * L) * LN2F;   // = -d^2 * log2(e), softmax score in exp2 domain
      }
      {
        const float diff2 = fmaxf(qn_m[i] + kn1 - 2.f * c1[i], 0.f);
        const float rden  = __builtin_amdgcn_rcpf(fmaxf(q1_m[i] * kc1, 1e-7f));
        float arg = fmaf(diff2 + diff2, rden, 1.f);
        arg = fmaxf(arg, 1.f + 1e-7f);
        const float L = __log2f(arg + __fsqrt_rn(arg * arg - 1.f));
        s1[i] = -(L * L) * LN2F;
      }
    }

    // ============== online softmax, exp2 domain (row m=i+hi8) ===============
    float rmx[8];
    #pragma unroll
    for (int i = 0; i < 8; ++i) rmx[i] = fmaxf(s0[i], s1[i]);
    #pragma unroll
    for (int off = 1; off < 16; off <<= 1)
      #pragma unroll
      for (int i = 0; i < 8; ++i) rmx[i] = fmaxf(rmx[i], __shfl_xor(rmx[i], off, 32));

    float p0[8], p1[8], rs[8], sc[8];
    #pragma unroll
    for (int i = 0; i < 8; ++i) {
      const float mn = fmaxf(mrow[i], rmx[i]);
      sc[i]  = __builtin_amdgcn_exp2f(mrow[i] - mn);   // native v_exp_f32 (base-2)
      p0[i]  = __builtin_amdgcn_exp2f(s0[i] - mn);
      p1[i]  = __builtin_amdgcn_exp2f(s1[i] - mn);
      rs[i]  = p0[i] + p1[i];
      mrow[i] = mn;
    }
    #pragma unroll
    for (int off = 1; off < 16; off <<= 1)
      #pragma unroll
      for (int i = 0; i < 8; ++i) rs[i] += __shfl_xor(rs[i], off, 32);
    #pragma unroll
    for (int i = 0; i < 8; ++i) {
      lrow[i] = lrow[i] * sc[i] + rs[i];
      o0[i] *= sc[i]; o1[i] *= sc[i]; o2[i] *= sc[i]; o3[i] *= sc[i];
    }

    // ===== P (C-layout) -> per-wave LDS pad -> A-layout; O += P * V =========
    // Same-wave DS ops complete in order, so no block barrier is needed here.
    _Float16* pw = &p_s[wave * 16 * P_STRIDE];
    #pragma unroll
    for (int i = 0; i < 8; ++i) {
      pw[(i + hi8) * P_STRIDE + col]      = (_Float16)p0[i];
      pw[(i + hi8) * P_STRIDE + 16 + col] = (_Float16)p1[i];
    }

    V16 pa;
    const _Float16* pp = &pw[col * P_STRIDE + hi8];
    pa.u[0] = *(const uint4*)(pp);
    pa.u[1] = *(const uint4*)(pp + 16);

    {
      V16 vb;
      const _Float16* vt = vt_s[cur];
      const _Float16* vp0 = &vt[(0  + col) * VT_STRIDE + hi8];
      vb.u[0] = *(const uint4*)(vp0); vb.u[1] = *(const uint4*)(vp0 + 16);
      o0 = wmma_f16(pa, vb, o0);
      const _Float16* vp1 = &vt[(16 + col) * VT_STRIDE + hi8];
      vb.u[0] = *(const uint4*)(vp1); vb.u[1] = *(const uint4*)(vp1 + 16);
      o1 = wmma_f16(pa, vb, o1);
      const _Float16* vp2 = &vt[(32 + col) * VT_STRIDE + hi8];
      vb.u[0] = *(const uint4*)(vp2); vb.u[1] = *(const uint4*)(vp2 + 16);
      o2 = wmma_f16(pa, vb, o2);
      const _Float16* vp3 = &vt[(48 + col) * VT_STRIDE + hi8];
      vb.u[0] = *(const uint4*)(vp3); vb.u[1] = *(const uint4*)(vp3 + 16);
      o3 = wmma_f16(pa, vb, o3);
    }
  }

  // ======================= normalize and write out ==========================
  #pragma unroll
  for (int i = 0; i < 8; ++i) {
    const float inv = 1.f / lrow[i];
    float* op = Og + (size_t)(q0 + wave * 16 + i + hi8) * D_DIM;
    op[col]      = o0[i] * inv;
    op[16 + col] = o1[i] * inv;
    op[32 + col] = o2[i] * inv;
    op[48 + col] = o3[i] * inv;
  }
}

extern "C" void kernel_launch(void* const* d_in, const int* in_sizes, int n_in,
                              void* d_out, int out_size, void* d_ws, size_t ws_size,
                              hipStream_t stream) {
  (void)in_sizes; (void)n_in; (void)out_size; (void)d_ws; (void)ws_size;
  const float* Q = (const float*)d_in[0];
  const float* K = (const float*)d_in[1];
  const float* V = (const float*)d_in[2];
  float* O = (float*)d_out;
  dim3 grid(BH * (S_LEN / QTILE));   // 32 * 16 = 512 blocks
  dim3 block(256);                   // 8 wave32s
  poincare_flash_attn<<<grid, block, 0, stream>>>(Q, K, V, O);
}
